// LAVAMemory_7370163879940
// MI455X (gfx1250) — matrix operand order repaired and checked by reading, not compiled, not run on hardware
//
#include <hip/hip_runtime.h>
#include <hip/hip_bf16.h>
#include <cstdint>

// ---------------------------------------------------------------------------
// LAVA memory read, MI455X (gfx1250, wave32, WMMA bf16, async global->LDS)
// Pipeline:
//   1. maxabs[h]  = max_n |contents[n,h]|            (atomic fp-as-uint max)
//   2. invn[n]    = 1/max(||addresses[n,:]||, eps)
//   3. query      = x @ W_addr^T                     (WMMA bf16)
//   4. qn (bf16)  = rownorm(query)
//   5. scores + running per-token top-8 per slab     (WMMA bf16,
//        B tile staged via global_load_async_to_lds_b128, double-buffered)
//   6. merge 32 partial top-8 lists -> softmax weights
//   7. gather 8 rows/token, int8 dequant, weighted sum
//   8. out        = read @ W_read^T                  (WMMA bf16)
// ---------------------------------------------------------------------------

typedef __bf16 bf16_t;
typedef __attribute__((ext_vector_type(16))) __bf16 v16bf;
typedef __attribute__((ext_vector_type(8)))  __bf16 v8bf;
typedef __attribute__((ext_vector_type(8)))  float  v8f;

constexpr int   kBdim = 4, kSdim = 512, kH = 1024;
constexpr int   kBS = kBdim * kSdim;            // 2048 tokens
constexpr int   kN = 65536;                     // latent slots
constexpr int   kSlabs = 32;                    // score slabs
constexpr int   kSlotsPerSlab = kN / kSlabs;    // 2048
constexpr int   kTokTile = 256;                 // tokens per scores workgroup
constexpr int   kSliceK = 256;                  // K per staged LDS slice
constexpr int   kRowPad = 260;                  // 256 + 4 floats (bank skew)
constexpr float kEpsNorm = 1e-8f;
constexpr float kEpsScale = 1e-6f;

// ---------------------------------------------------------------------------
// Fragment loaders (wave32 WMMA 16x16x32 bf16 layouts as used here (A and B: contiguous k))
// A (16x32, 16-bit): lanes 0-15 -> M=0..15, K = 0..15 contiguous;
//                    lanes 16-31 -> M=0..15, K = 16..31 contiguous
// B (32x16, 16-bit): lane n = lane&15, K = (lane>>4)*16 + {0..15} contiguous
// D (16x16 f32):     n = lane&15, m = r + 8*(lane>>4)
// ---------------------------------------------------------------------------

__device__ __forceinline__ v8f wmma_bf16(v16bf a, v16bf b, v8f c) {
  return __builtin_amdgcn_wmma_f32_16x16x32_bf16(
      /*neg_a=*/false, a, /*neg_b=*/false, b,
      /*c_mod=*/(short)0, c, /*reuse_a=*/false, /*reuse_b=*/false);
}

__device__ __forceinline__ v8f zero_v8f() { v8f z = {}; return z; }

// p = &A[m*K + kk + (lane>>4)*16], f32 source, convert to bf16
__device__ __forceinline__ v16bf load_a_f32(const float* p) {
  float lo[8], hi[8];
  *(float4*)&lo[0] = *(const float4*)(p);
  *(float4*)&lo[4] = *(const float4*)(p + 4);
  *(float4*)&hi[0] = *(const float4*)(p + 8);
  *(float4*)&hi[4] = *(const float4*)(p + 12);
  v16bf a;
#pragma unroll
  for (int j = 0; j < 8; ++j) { a[j] = (__bf16)lo[j]; a[j + 8] = (__bf16)hi[j]; }
  return a;
}

// p = &A[m*K + kk + (lane>>4)*16], bf16 source (L2-resident q_norm)
__device__ __forceinline__ v16bf load_a_bf16(const bf16_t* p) {
  v8bf lo = *(const v8bf*)(p);
  v8bf hi = *(const v8bf*)(p + 8);
  v16bf a;
#pragma unroll
  for (int j = 0; j < 8; ++j) { a[j] = lo[j]; a[j + 8] = hi[j]; }
  return a;
}

// p = &B[n*K + kk + (lane>>4)*16] (global or LDS), f32 scaled by s
__device__ __forceinline__ v16bf load_b_f32(const float* p, float s) {
  float b[16];
  *(float4*)&b[0]  = *(const float4*)(p);
  *(float4*)&b[4]  = *(const float4*)(p + 4);
  *(float4*)&b[8]  = *(const float4*)(p + 8);
  *(float4*)&b[12] = *(const float4*)(p + 12);
  v16bf r;
#pragma unroll
  for (int j = 0; j < 16; ++j) r[j] = (__bf16)(b[j] * s);
  return r;
}

__device__ __forceinline__ void wait_async0() {
  asm volatile("s_wait_asynccnt 0x0" ::: "memory");
}

// ---------------------------------------------------------------------------
// 1. per-channel max|contents| (columns of [N,H]); coalesced along h
// ---------------------------------------------------------------------------
__global__ __launch_bounds__(256) void maxabs_kernel(
    const float* __restrict__ contents, float* __restrict__ maxabs) {
  const int h = blockIdx.x * 256 + threadIdx.x;
  const int n0 = blockIdx.y * (kN / 64);
  float m = 0.f;
  for (int n = n0; n < n0 + kN / 64; ++n)
    m = fmaxf(m, fabsf(contents[(size_t)n * kH + h]));
  atomicMax((unsigned int*)&maxabs[h], __float_as_uint(m));  // nonneg fp == uint order
}

// ---------------------------------------------------------------------------
// 2. address inverse row norms: one wave (32 lanes) per row
// ---------------------------------------------------------------------------
__global__ __launch_bounds__(256) void addr_inv_norm(
    const float* __restrict__ addr, float* __restrict__ invn) {
  const int lane = threadIdx.x & 31;
  const int row = blockIdx.x * 8 + (threadIdx.x >> 5);
  const float* p = addr + (size_t)row * kH;
  float ss = 0.f;
  for (int i = 0; i < kH / 32; ++i) { float v = p[i * 32 + lane]; ss += v * v; }
#pragma unroll
  for (int off = 16; off > 0; off >>= 1) ss += __shfl_xor(ss, off);
  if (lane == 0) invn[row] = 1.0f / fmaxf(sqrtf(ss), kEpsNorm);
}

// ---------------------------------------------------------------------------
// 3 & 8. GEMM  C[M,Nout] = A[M,K] * B[Nout,K]^T  (f32 in, bf16 WMMA, f32 out)
//   grid: (Nout/16, M/128), 8 waves/block, one 16x16 tile per wave
// ---------------------------------------------------------------------------
__global__ __launch_bounds__(256) void gemm_at_bt(
    const float* __restrict__ A, const float* __restrict__ Bm,
    float* __restrict__ C, int K, int Nout) {
  const int lane = threadIdx.x & 31;
  const int wv = threadIdx.x >> 5;
  const int m0 = blockIdx.y * 128 + wv * 16;
  const int n0 = blockIdx.x * 16;
  const int kb8 = (lane >> 4) * 16;
  const int kb16 = (lane >> 4) * 16;
  const float* arow = A + (size_t)(m0 + (lane & 15)) * K;
  const float* brow = Bm + (size_t)(n0 + (lane & 15)) * K;
  v8f acc = {};
  for (int kk = 0; kk < K; kk += 32) {
    v16bf a = load_a_f32(arow + kk + kb8);
    v16bf b = load_b_f32(brow + kk + kb16, 1.0f);
    acc = wmma_bf16(a, b, acc);
  }
  const int cn = n0 + (lane & 15);
  const int cm = m0 + 8 * (lane >> 4);
#pragma unroll
  for (int r = 0; r < 8; ++r) C[(size_t)(cm + r) * Nout + cn] = acc[r];
}

// ---------------------------------------------------------------------------
// 4. row-normalize query -> bf16 (one wave per row)
// ---------------------------------------------------------------------------
__global__ __launch_bounds__(256) void row_norm_bf16(
    const float* __restrict__ q, bf16_t* __restrict__ qn) {
  const int lane = threadIdx.x & 31;
  const int row = blockIdx.x * 8 + (threadIdx.x >> 5);
  const float* p = q + (size_t)row * kH;
  float ss = 0.f;
  for (int i = 0; i < kH / 32; ++i) { float v = p[i * 32 + lane]; ss += v * v; }
#pragma unroll
  for (int off = 16; off > 0; off >>= 1) ss += __shfl_xor(ss, off);
  const float inv = 1.0f / fmaxf(sqrtf(ss), kEpsNorm);
  for (int i = 0; i < kH / 32; ++i)
    qn[(size_t)row * kH + i * 32 + lane] = (bf16_t)(p[i * 32 + lane] * inv);
}

// ---------------------------------------------------------------------------
// 5. scores + per-slab running top-8.
//    grid (kSlabs, kBS/256); 8 waves; wave w owns tokens m0 = base + 32w,
//    computing two 16x16 WMMA tiles (subtiles 0/1) per 16-slot group.
//    The 16-slot address tile is staged to LDS in 256-wide K slices with
//    global_load_async_to_lds_b128 (double-buffered; async overlaps WMMA),
//    normalized on the fly (invn fused into the bf16 convert at consume).
//    Each lane owns exactly one token's top-8 list in registers.
// ---------------------------------------------------------------------------
__global__ __launch_bounds__(256) void scores_topk(
    const bf16_t* __restrict__ qn, const float* __restrict__ addr,
    const float* __restrict__ invn, float* __restrict__ pvals,
    int* __restrict__ pidx) {
  __shared__ float bstage[2][16][kRowPad];  // 2 x 16 slots x 256K (+pad) ~32.5 KB
  __shared__ float sc[8][2][16][16];        // [wave][subtile][m][n] = 16 KB
  const int lane = threadIdx.x & 31;
  const int wv = threadIdx.x >> 5;
  const int slab = blockIdx.x;
  const int slotBase = slab * kSlotsPerSlab;
  const int m0 = blockIdx.y * kTokTile + wv * 32;
  const int kb8 = (lane >> 4) * 16;
  const int kb16 = (lane >> 4) * 16;
  const bf16_t* aro0 = qn + (size_t)(m0 + (lane & 15)) * kH;
  const bf16_t* aro1 = qn + (size_t)(m0 + 16 + (lane & 15)) * kH;

  // stage slice u (16 slots x 256 K f32) into LDS buffer b, async
  auto stage = [&](int u, int b) {
#pragma unroll
    for (int t = 0; t < 4; ++t) {
      const int q = threadIdx.x + t * 256;      // 0..1023 float4's
      const int row = q >> 6;                   // 0..15 (slot in group)
      const int c4 = (q & 63) << 2;             // float column 0..252
      const float* gsrc = addr +
          (size_t)(slotBase + (u >> 2) * 16 + row) * kH + (u & 3) * kSliceK + c4;
      const uint32_t ldsa = (uint32_t)(uintptr_t)&bstage[b][row][c4];
      asm volatile("global_load_async_to_lds_b128 %0, %1, off"
                   :: "v"(ldsa), "v"(gsrc) : "memory");
    }
  };

  float tv[8]; int ti[8];
#pragma unroll
  for (int j = 0; j < 8; ++j) { tv[j] = -1e30f; ti[j] = slotBase; }
  float curMin = -1e30f;

  constexpr int kTotalSlices = (kSlotsPerSlab / 16) * (kH / kSliceK);  // 512

  stage(0, 0);
  wait_async0();
  __syncthreads();

  v8f acc0 = zero_v8f(), acc1 = zero_v8f();
  float sInv = 0.f;
  int n0 = slotBase;

  for (int u = 0; u < kTotalSlices; ++u) {
    const int g = u >> 2, s = u & 3, b = u & 1;
    if (s == 0) {
      n0 = slotBase + g * 16;
      sInv = invn[n0 + (lane & 15)];
      acc0 = zero_v8f();
      acc1 = zero_v8f();
    }
    if (u + 1 < kTotalSlices) stage(u + 1, (u + 1) & 1);  // overlap with WMMA

    const int kkg = s * kSliceK;
    for (int kk = 0; kk < kSliceK; kk += 32) {
      v16bf bfr = load_b_f32(&bstage[b][lane & 15][kk + kb16], sInv);
      v16bf a0 = load_a_bf16(aro0 + kkg + kk + kb8);
      v16bf a1 = load_a_bf16(aro1 + kkg + kk + kb8);
      acc0 = wmma_bf16(a0, bfr, acc0);
      acc1 = wmma_bf16(a1, bfr, acc1);
    }
    wait_async0();      // this wave's async stores for slice u+1 landed
    __syncthreads();    // all waves' stores landed; all reads of buf b done

    if (s == 3) {
      // dump D tiles: n = lane&15, m = r + 8*(lane>>4)
#pragma unroll
      for (int r = 0; r < 8; ++r) {
        sc[wv][0][r + 8 * (lane >> 4)][lane & 15] = acc0[r];
        sc[wv][1][r + 8 * (lane >> 4)][lane & 15] = acc1[r];
      }
      __syncthreads();
      // lane owns token m0+lane: subtile = lane>>4, row = lane&15
      const int st = lane >> 4, row = lane & 15;
#pragma unroll
      for (int c = 0; c < 16; ++c) {
        const float v = sc[wv][st][row][c];
        if (v > curMin) {                       // fast reject: rarely true
          int mp = 0; float mv = tv[0];
#pragma unroll
          for (int j = 1; j < 8; ++j) if (tv[j] < mv) { mv = tv[j]; mp = j; }
#pragma unroll
          for (int j = 0; j < 8; ++j) {
            const bool sel = (j == mp);
            tv[j] = sel ? v : tv[j];
            ti[j] = sel ? (n0 + c) : ti[j];
          }
          mv = tv[0];
#pragma unroll
          for (int j = 1; j < 8; ++j) mv = fminf(mv, tv[j]);
          curMin = mv;
        }
      }
      __syncthreads();
    }
  }

  const int tok = m0 + lane;
#pragma unroll
  for (int j = 0; j < 8; ++j) {
    pvals[((size_t)tok * kSlabs + slab) * 8 + j] = tv[j];
    pidx[((size_t)tok * kSlabs + slab) * 8 + j] = ti[j];
  }
}

// ---------------------------------------------------------------------------
// 6. merge 32 partial top-8 lists per token (one wave per token, lane = slab),
//    then softmax over the final 8.
// ---------------------------------------------------------------------------
__global__ __launch_bounds__(256) void merge_topk(
    const float* __restrict__ pvals, const int* __restrict__ pidx,
    float* __restrict__ wout, int* __restrict__ iout) {
  const int lane = threadIdx.x & 31;
  const int t = blockIdx.x * 8 + (threadIdx.x >> 5);
  float v[8]; int ix[8];
#pragma unroll
  for (int j = 0; j < 8; ++j) {
    v[j]  = pvals[(size_t)t * (kSlabs * 8) + lane * 8 + j];
    ix[j] = pidx[(size_t)t * (kSlabs * 8) + lane * 8 + j];
  }
  float selv[8]; int seli[8];
#pragma unroll
  for (int it = 0; it < 8; ++it) {
    float bv = v[0]; int bj = 0; int bix = ix[0];
#pragma unroll
    for (int j = 1; j < 8; ++j)
      if (v[j] > bv) { bv = v[j]; bj = j; bix = ix[j]; }
    int bkey = (lane << 3) | bj;
#pragma unroll
    for (int off = 16; off > 0; off >>= 1) {
      const float ov = __shfl_xor(bv, off);
      const int os = __shfl_xor(bix, off);
      const int ok = __shfl_xor(bkey, off);
      if (ov > bv || (ov == bv && ok < bkey)) { bv = ov; bix = os; bkey = ok; }
    }
    selv[it] = bv; seli[it] = bix;
    const bool win = ((bkey >> 3) == lane);
    const int wj = bkey & 7;
#pragma unroll
    for (int j = 0; j < 8; ++j)
      if (win && j == wj) v[j] = -1e30f;
  }
  const float mx = selv[0];
  float ev[8], sum = 0.f;
#pragma unroll
  for (int k = 0; k < 8; ++k) { ev[k] = expf(selv[k] - mx); sum += ev[k]; }
  const float inv = 1.0f / sum;
  if (lane == 0) {
#pragma unroll
    for (int k = 0; k < 8; ++k) {
      wout[(size_t)t * 8 + k] = ev[k] * inv;
      iout[(size_t)t * 8 + k] = seli[k];
    }
  }
}

// ---------------------------------------------------------------------------
// 7. gather 8 rows per token, int8 per-channel dequant, weighted sum
// ---------------------------------------------------------------------------
__global__ __launch_bounds__(256) void gather_read(
    const float* __restrict__ contents, const float* __restrict__ maxabs,
    const float* __restrict__ wsel, const int* __restrict__ isel,
    float* __restrict__ readv) {
  const int t = blockIdx.x;
  const int tid = threadIdx.x;
  float wk[8]; int ik[8];
#pragma unroll
  for (int k = 0; k < 8; ++k) {
    wk[k] = wsel[(size_t)t * 8 + k];
    ik[k] = isel[(size_t)t * 8 + k];
  }
  float scl[4], acc[4];
#pragma unroll
  for (int j = 0; j < 4; ++j) {
    scl[j] = fmaxf(maxabs[tid + j * 256] * (1.0f / 127.0f), kEpsScale);
    acc[j] = 0.f;
  }
#pragma unroll
  for (int k = 0; k < 8; ++k) {
    const float* crow = contents + (size_t)ik[k] * kH;
#pragma unroll
    for (int j = 0; j < 4; ++j) {
      const float c = crow[tid + j * 256];
      const float q =
          fminf(fmaxf(rintf(c / scl[j]), -127.f), 127.f) * scl[j];
      acc[j] += wk[k] * q;
    }
  }
#pragma unroll
  for (int j = 0; j < 4; ++j)
    readv[(size_t)t * kH + tid + j * 256] = acc[j];
}

// ---------------------------------------------------------------------------
extern "C" void kernel_launch(void* const* d_in, const int* in_sizes, int n_in,
                              void* d_out, int out_size, void* d_ws,
                              size_t ws_size, hipStream_t stream) {
  (void)in_sizes; (void)n_in; (void)out_size; (void)ws_size;
  const float* x     = (const float*)d_in[0];   // [B,S,H]
  const float* addr  = (const float*)d_in[1];   // [N,H]
  const float* cont  = (const float*)d_in[2];   // [N,H]
  const float* Waddr = (const float*)d_in[3];   // [H,H]
  const float* Wread = (const float*)d_in[4];   // [H,H]
  float* out = (float*)d_out;                   // [B,S,H]

  char* ws = (char*)d_ws;
  size_t off = 0;
  auto alloc = [&](size_t bytes) {
    size_t o = off;
    off = (off + bytes + 255) & ~(size_t)255;
    return o;
  };
  float*  maxabs = (float*)(ws + alloc((size_t)kH * 4));
  float*  invn   = (float*)(ws + alloc((size_t)kN * 4));
  float*  query  = (float*)(ws + alloc((size_t)kBS * kH * 4));
  bf16_t* qn     = (bf16_t*)(ws + alloc((size_t)kBS * kH * 2));
  float*  pvals  = (float*)(ws + alloc((size_t)kBS * kSlabs * 8 * 4));
  int*    pidx   = (int*)(ws + alloc((size_t)kBS * kSlabs * 8 * 4));
  float*  wsel   = (float*)(ws + alloc((size_t)kBS * 8 * 4));
  int*    isel   = (int*)(ws + alloc((size_t)kBS * 8 * 4));
  float*  readv  = (float*)(ws + alloc((size_t)kBS * kH * 4));

  hipMemsetAsync(maxabs, 0, kH * sizeof(float), stream);
  maxabs_kernel<<<dim3(kH / 256, 64), 256, 0, stream>>>(cont, maxabs);
  addr_inv_norm<<<kN / 8, 256, 0, stream>>>(addr, invn);
  gemm_at_bt<<<dim3(kH / 16, kBS / 128), 256, 0, stream>>>(x, Waddr, query, kH, kH);
  row_norm_bf16<<<kBS / 8, 256, 0, stream>>>(query, qn);
  scores_topk<<<dim3(kSlabs, kBS / kTokTile), 256, 0, stream>>>(qn, addr, invn,
                                                                pvals, pidx);
  merge_topk<<<kBS / 8, 256, 0, stream>>>(pvals, pidx, wsel, isel);
  gather_read<<<kBS, 256, 0, stream>>>(cont, maxabs, wsel, isel, readv);
  gemm_at_bt<<<dim3(kH / 16, kBS / 128), 256, 0, stream>>>(readv, Wread, out, kH, kH);
}
